// SingleSelfAttnGRU_74122545594412
// MI455X (gfx1250) — matrix operand order repaired and checked by reading, not compiled, not run on hardware
//
#include <hip/hip_runtime.h>
#include <hip/hip_bf16.h>
#include <stdint.h>

#define SEQ   512
#define BATCH 256
#define ISZ   128
#define HSZ   128
#define G3    384       // 3*H
#define AWAIT 3

typedef _Float16 h16;
typedef __attribute__((ext_vector_type(16))) _Float16 v16h;
typedef __attribute__((ext_vector_type(8)))  float    v8f;

// ---------------------------------------------------------------------------
// WMMA helpers (CDNA5 V_WMMA_F32_16X16X32_F16, wave32)
// ---------------------------------------------------------------------------
__device__ inline v8f wmma_f16(v16h a, v16h b, v8f c) {
  return __builtin_amdgcn_wmma_f32_16x16x32_f16(false, a, false, b, (short)0, c,
                                                false, false);
}

// A fragment (16x32, f16) built from row-major source with leading dim `ld`.
// k order used here (contiguous, same as B): lanes 0-15 -> M=lane, K = kc*32 + 0..15
//             lanes 16-31 -> M=lane-16, K = kc*32 + 16..31
template <typename PT>
__device__ inline v16h ldA(const PT* base, int lane, int kc, int ld) {
  const int m  = lane & 15;
  const int ks = (lane < 16) ? 0 : 16;
  const PT* p  = base + m * ld + kc * 32 + ks;
  v16h a;
#pragma unroll
  for (int i = 0; i < 8; ++i) a[i] = (h16)p[i];
#pragma unroll
  for (int i = 0; i < 8; ++i) a[8 + i] = (h16)p[8 + i];
  return a;
}

// B fragment (32x16) where B[k][n] = W[n0+n][koff+k], W row-major, ld = row len.
// Contiguous k order (as ldA): lanes 0-15 -> N=lane, K=kc*32+0..15 ; lanes 16-31 -> K=kc*32+16..31
template <typename PT>
__device__ inline v16h ldBrow(const PT* w, int lane, int n0, int kc, int ld,
                              int koff = 0) {
  const int n  = n0 + (lane & 15);
  const int kb = kc * 32 + ((lane < 16) ? 0 : 16) + koff;
  const PT* p  = w + n * ld + kb;
  v16h b;
#pragma unroll
  for (int i = 0; i < 16; ++i) b[i] = (h16)p[i];
  return b;
}

// B fragment where B[k][n] = W[k][n0+n] (no transpose), strided reads.
__device__ inline v16h ldBcol(const float* w, int lane, int n0, int kc, int ld) {
  const int n  = n0 + (lane & 15);
  const int kb = kc * 32 + ((lane < 16) ? 0 : 16);
  v16h b;
#pragma unroll
  for (int i = 0; i < 16; ++i) b[i] = (h16)w[(kb + i) * ld + n];
  return b;
}

// ---------------------------------------------------------------------------
// gfx1250 async copy global -> LDS (ASYNCcnt-tracked)
// ---------------------------------------------------------------------------
__device__ inline void async_ld16(uint32_t lds_off, uint64_t gaddr) {
  asm volatile("global_load_async_to_lds_b128 %0, %1, off"
               :: "v"(lds_off), "v"(gaddr) : "memory");
}

// ---------------------------------------------------------------------------
// Kernel 1: time-parallel input projections.
//   gid[t,b,:] = x[t,b,:] @ w_ih_dec[:, :128].T + b_ih_dec   (f16 out)
//   gie[t,b,:] = x[t,b,:] @ w_ih_enc.T          + b_ih_enc   (f16 out)
// One block per 16 flattened (t,b) rows; 4 waves x 12 tile chains x 4 WMMAs.
// ---------------------------------------------------------------------------
__global__ __launch_bounds__(128) void gi_precompute_kernel(
    const float* __restrict__ x, const float* __restrict__ wihd,
    const float* __restrict__ wihe, const float* __restrict__ bihd,
    const float* __restrict__ bihe, h16* __restrict__ gid,
    h16* __restrict__ gie) {
  const int tid  = threadIdx.x;
  const int lane = tid & 31;
  const int wv   = tid >> 5;        // 0..3
  const int mt   = blockIdx.x;      // rows mt*16 .. mt*16+15 of (t*256+b)

  v16h a[4];
#pragma unroll
  for (int kc = 0; kc < 4; ++kc)
    a[kc] = ldA(x + (size_t)mt * 16 * ISZ, lane, kc, ISZ);

  const int col = lane & 15;
  const int mh  = (lane >> 4) << 3;

  for (int c = wv; c < 48; c += 4) {
    const int mat = c & 1;          // 0 = dec, 1 = enc
    const int n0  = (c >> 1) * 16;
    const float* W  = mat ? wihe : wihd;
    const int    ld = mat ? ISZ : (ISZ + HSZ);
    const float* bi = mat ? bihe : bihd;
    h16* out        = mat ? gie : gid;

    const float bias = bi[n0 + col];
    v8f acc;
#pragma unroll
    for (int j = 0; j < 8; ++j) acc[j] = bias;
#pragma unroll
    for (int kc = 0; kc < 4; ++kc)
      acc = wmma_f16(a[kc], ldBrow(W, lane, n0, kc, ld), acc);
#pragma unroll
    for (int j = 0; j < 8; ++j) {
      const int row = mt * 16 + j + mh;
      out[(size_t)row * G3 + n0 + col] = (h16)acc[j];
    }
  }
}

// ---------------------------------------------------------------------------
// Kernel 2: persistent sequential recurrence. 16 blocks x 256 threads;
// block g owns batch rows [g*16, g*16+16). Recurrent weights live in LDS (f16),
// gi[t] double-buffered into LDS via async-to-LDS, overlapped with WMMAs.
// LDS layout (bytes):
#define OFF_WHD 0                       // w_hh_dec f16  [384][128]  96KB
#define SZ_WHD  (G3 * HSZ * 2)
#define OFF_WHE (OFF_WHD + SZ_WHD)      // w_hh_enc f16              96KB
#define OFF_GI  (OFF_WHE + SZ_WHD)      // gi ping-pong [2][dec|enc][16][384] f16
#define SZ_GIB  (16 * G3 * 2 * 2)       // 24576 B per buffer
#define OFF_P   (OFF_GI + 2 * SZ_GIB)   // P[3][16][384] f16         36KB
#define SZ_P    (3 * 16 * G3 * 2)
#define OFF_V   (OFF_P + SZ_P)          // V[3][16][128] f16         12KB
#define SZ_V    (3 * 16 * HSZ * 2)
#define OFF_DH  (OFF_V + SZ_V)          // dh f32 [16][128]           8KB
#define OFF_EH  (OFF_DH + 16 * HSZ * 4) // eh f32                     8KB
#define OFF_BHD (OFF_EH + 16 * HSZ * 4) // b_hh_dec f32
#define OFF_BHE (OFF_BHD + G3 * 4)
#define OFF_BQ  (OFF_BHE + G3 * 4)
#define OFF_C3  (OFF_BQ + HSZ * 4)      // c[3][16]
#define OFF_SC  (OFF_C3 + 3 * 16 * 4)   // score partials [48][4]
#define OFF_S   (OFF_SC + 48 * 4 * 4)   // scores [3][16]
#define OFF_AW  (OFF_S + 3 * 16 * 4)    // softmax weights [3][16]
#define OFF_LL  (OFF_AW + 3 * 16 * 4)   // lengths [16]
#define SMEM_TOTAL (OFF_LL + 16 * 4)    // = 316288 B  (< 320KB WGP LDS)
// ---------------------------------------------------------------------------
__device__ inline void load_gi_async(uint32_t buf_lds, const h16* gid,
                                     const h16* gie, int t, int g16, int tid) {
  const uint64_t gd =
      (uint64_t)(uintptr_t)(gid + ((size_t)t * BATCH + g16) * G3);
  const uint64_t ge =
      (uint64_t)(uintptr_t)(gie + ((size_t)t * BATCH + g16) * G3);
  const int off = tid * 48;           // 256 threads * 48B = 12288B per half
#pragma unroll
  for (int q = 0; q < 3; ++q) {
    async_ld16(buf_lds + off + q * 16, gd + off + q * 16);
    async_ld16(buf_lds + 12288 + off + q * 16, ge + off + q * 16);
  }
}

__global__ __launch_bounds__(256) void SingleSelfAttnGRU_74122545594412_kernel(
    const h16* __restrict__ gid, const h16* __restrict__ gie,
    const float* __restrict__ whd, const float* __restrict__ whe,
    const float* __restrict__ bhd, const float* __restrict__ bhe,
    const float* __restrict__ wihd, const float* __restrict__ wq,
    const float* __restrict__ bq, const int* __restrict__ length,
    const float* __restrict__ eh0, const float* __restrict__ dh0,
    float* __restrict__ out) {
  extern __shared__ __align__(16) char smem[];
  h16*   WHD = (h16*)(smem + OFF_WHD);
  h16*   WHE = (h16*)(smem + OFF_WHE);
  h16*   GIB = (h16*)(smem + OFF_GI);
  h16*   P   = (h16*)(smem + OFF_P);
  h16*   V   = (h16*)(smem + OFF_V);
  float* DH  = (float*)(smem + OFF_DH);
  float* EH  = (float*)(smem + OFF_EH);
  float* BHD = (float*)(smem + OFF_BHD);
  float* BHE = (float*)(smem + OFF_BHE);
  float* BQ  = (float*)(smem + OFF_BQ);
  float* C3  = (float*)(smem + OFF_C3);
  float* SC  = (float*)(smem + OFF_SC);
  float* S   = (float*)(smem + OFF_S);
  float* AW  = (float*)(smem + OFF_AW);
  int*   LL  = (int*)(smem + OFF_LL);

  const int tid  = threadIdx.x;
  const int lane = tid & 31;
  const int wv   = tid >> 5;                 // 0..7
  const int g16  = blockIdx.x * 16;
  const int col  = lane & 15;
  const int mh   = (lane >> 4) << 3;
  const int w16  = wv * 16;

  // ---- setup: stage recurrent weights (f32->f16), biases, init states ----
  for (int i = tid; i < G3 * HSZ; i += 256) {
    WHD[i] = (h16)whd[i];
    WHE[i] = (h16)whe[i];
  }
  for (int i = tid; i < G3; i += 256) { BHD[i] = bhd[i]; BHE[i] = bhe[i]; }
  for (int i = tid; i < HSZ; i += 256) BQ[i] = bq[i];
  for (int i = tid; i < 16 * HSZ; i += 256) {
    DH[i] = dh0[i & 127];
    EH[i] = eh0[i & 127];
  }
  if (tid < 16) LL[tid] = length[g16 + tid];

  const uint32_t gi_lds0 = (uint32_t)(uintptr_t)GIB;
  load_gi_async(gi_lds0, gid, gie, 0, g16, tid);   // prime t=0
  __syncthreads();

  for (int t = 0; t < SEQ; ++t) {
    // A: prefetch gi[t+1] into the other buffer (ASYNCcnt-tracked)
    if (t + 1 < SEQ)
      load_gi_async(gi_lds0 + ((t + 1) & 1) * SZ_GIB, gid, gie, t + 1, g16, tid);

    // B: attention weights (hbuf frozen; score_j = dh.V_j + c_j)
    if (t > AWAIT) {
      const int d = tid >> 2, p = tid & 3;
      if (d < 48) {
        const int j = d >> 4, b = d & 15;
        const h16*   vp = V + (j * 16 + b) * HSZ + p * 32;
        const float* hp = DH + b * HSZ + p * 32;
        float s = 0.f;
#pragma unroll
        for (int k = 0; k < 32; ++k) s += hp[k] * (float)vp[k];
        SC[d * 4 + p] = s;
      }
      __syncthreads();
      if (tid < 48) {
        const int j = tid >> 4, b = tid & 15;
        S[tid] = SC[tid * 4] + SC[tid * 4 + 1] + SC[tid * 4 + 2] +
                 SC[tid * 4 + 3] + C3[j * 16 + b];
      }
      __syncthreads();
      if (tid < 16) {
        float s0 = S[tid], s1 = S[16 + tid], s2 = S[32 + tid];
        float mx = fmaxf(s0, fmaxf(s1, s2));
        float e0 = __expf(s0 - mx), e1 = __expf(s1 - mx), e2 = __expf(s2 - mx);
        float inv = 1.f / (e0 + e1 + e2);
        AW[tid] = e0 * inv; AW[16 + tid] = e1 * inv; AW[32 + tid] = e2 * inv;
      }
    }

    // C: recurrent matmuls  gh = h @ W_hh.T + b_hh  (24 WMMAs per wave)
    v16h adh[4], aeh[4];
#pragma unroll
    for (int kc = 0; kc < 4; ++kc) {
      adh[kc] = ldA(DH, lane, kc, HSZ);
      aeh[kc] = ldA(EH, lane, kc, HSZ);
    }
    v8f dr, dz, dn, er, ez, en;
#pragma unroll
    for (int j = 0; j < 8; ++j) {
      dr[j] = BHD[w16 + col]; dz[j] = BHD[128 + w16 + col]; dn[j] = BHD[256 + w16 + col];
      er[j] = BHE[w16 + col]; ez[j] = BHE[128 + w16 + col]; en[j] = BHE[256 + w16 + col];
    }
#pragma unroll
    for (int kc = 0; kc < 4; ++kc) {
      dr = wmma_f16(adh[kc], ldBrow(WHD, lane, w16, kc, HSZ), dr);
      dz = wmma_f16(adh[kc], ldBrow(WHD, lane, 128 + w16, kc, HSZ), dz);
      dn = wmma_f16(adh[kc], ldBrow(WHD, lane, 256 + w16, kc, HSZ), dn);
      er = wmma_f16(aeh[kc], ldBrow(WHE, lane, w16, kc, HSZ), er);
      ez = wmma_f16(aeh[kc], ldBrow(WHE, lane, 128 + w16, kc, HSZ), ez);
      en = wmma_f16(aeh[kc], ldBrow(WHE, lane, 256 + w16, kc, HSZ), en);
    }

    // D: wait for this step's gi (keep t+1's 6 ops in flight), then sync
    if (t + 1 < SEQ) asm volatile("s_wait_asynccnt 0x6" ::: "memory");
    else             asm volatile("s_wait_asynccnt 0x0" ::: "memory");
    __syncthreads();

    // E: gating (wave w owns gate columns w16..w16+15 of each gate group)
    const h16* gdb = GIB + (t & 1) * 12288;  // dec gi [16][384] (h16 units)
    const h16* geb = gdb + 6144;             // enc gi [16][384]
    const bool use_ctx = (t > AWAIT);
    float dnew[8], enew[8];
#pragma unroll
    for (int j = 0; j < 8; ++j) {
      const int M = j + mh;
      const float msk = (LL[M] > t) ? 1.f : 0.f;
      float gr = (float)gdb[M * G3 + w16 + col];
      float gz = (float)gdb[M * G3 + 128 + w16 + col];
      float gn = (float)gdb[M * G3 + 256 + w16 + col];
      if (use_ctx) {  // ctx contribution: sum_j aw_j * P_j
        const float a0 = AW[M], a1 = AW[16 + M], a2 = AW[32 + M];
        const h16* p0 = P + (0 * 16 + M) * G3;
        const h16* p1 = P + (1 * 16 + M) * G3;
        const h16* p2 = P + (2 * 16 + M) * G3;
        gr += a0 * (float)p0[w16 + col] + a1 * (float)p1[w16 + col] + a2 * (float)p2[w16 + col];
        gz += a0 * (float)p0[128 + w16 + col] + a1 * (float)p1[128 + w16 + col] + a2 * (float)p2[128 + w16 + col];
        gn += a0 * (float)p0[256 + w16 + col] + a1 * (float)p1[256 + w16 + col] + a2 * (float)p2[256 + w16 + col];
      }
      float r = 1.f / (1.f + __expf(-(gr + dr[j])));
      float z = 1.f / (1.f + __expf(-(gz + dz[j])));
      float n = tanhf(gn + r * dn[j]);
      dnew[j] = ((1.f - z) * n + z * DH[M * HSZ + w16 + col]) * msk;

      float r2 = 1.f / (1.f + __expf(-((float)geb[M * G3 + w16 + col] + er[j])));
      float z2 = 1.f / (1.f + __expf(-((float)geb[M * G3 + 128 + w16 + col] + ez[j])));
      float n2 = tanhf((float)geb[M * G3 + 256 + w16 + col] + r2 * en[j]);
      enew[j] = ((1.f - z2) * n2 + z2 * EH[M * HSZ + w16 + col]) * msk;
    }
#pragma unroll
    for (int j = 0; j < 8; ++j) {
      const int M = j + mh;
      DH[M * HSZ + w16 + col] = dnew[j];
      EH[M * HSZ + w16 + col] = enew[j];
    }
    __syncthreads();

    // G: hbuf slot t (t<3): precompute P_t = eh@Wc.T, V_t = eh@wq, c_t = bq.eh
    if (t < AWAIT) {
      v16h ae2[4];
#pragma unroll
      for (int kc = 0; kc < 4; ++kc) ae2[kc] = ldA(EH, lane, kc, HSZ);
#pragma unroll
      for (int s = 0; s < 3; ++s) {      // 24 P-tiles over 8 waves
        const int n0 = (wv + 8 * s) * 16;
        v8f acc;
#pragma unroll
        for (int j = 0; j < 8; ++j) acc[j] = 0.f;
#pragma unroll
        for (int kc = 0; kc < 4; ++kc)
          acc = wmma_f16(ae2[kc], ldBrow(wihd, lane, n0, kc, ISZ + HSZ, 128), acc);
#pragma unroll
        for (int j = 0; j < 8; ++j)
          P[(t * 16 + j + mh) * G3 + n0 + col] = (h16)acc[j];
      }
      {                                   // 8 V-tiles, one per wave
        v8f acc;
#pragma unroll
        for (int j = 0; j < 8; ++j) acc[j] = 0.f;
#pragma unroll
        for (int kc = 0; kc < 4; ++kc)
          acc = wmma_f16(ae2[kc], ldBcol(wq, lane, w16, kc, HSZ), acc);
#pragma unroll
        for (int j = 0; j < 8; ++j)
          V[(t * 16 + j + mh) * HSZ + w16 + col] = (h16)acc[j];
      }
      if (tid < 16) {
        float s = 0.f;
        for (int k = 0; k < HSZ; ++k) s += BQ[k] * EH[tid * HSZ + k];
        C3[t * 16 + tid] = s;
      }
      // no barrier needed: next consumers are separated by step-D barriers
    }
  }

  // final dh -> output (mask at t=511 makes this exactly zero, matching ref)
  for (int i = tid; i < 16 * HSZ; i += 256)
    out[(size_t)(g16 + (i >> 7)) * HSZ + (i & 127)] = DH[i];
}

// ---------------------------------------------------------------------------
extern "C" void kernel_launch(void* const* d_in, const int* in_sizes, int n_in,
                              void* d_out, int out_size, void* d_ws,
                              size_t ws_size, hipStream_t stream) {
  const float* x    = (const float*)d_in[0];
  const int*   len  = (const int*)d_in[1];
  const float* wihd = (const float*)d_in[2];
  const float* whd  = (const float*)d_in[3];
  const float* bihd = (const float*)d_in[4];
  const float* bhd  = (const float*)d_in[5];
  const float* wihe = (const float*)d_in[6];
  const float* whe  = (const float*)d_in[7];
  const float* bihe = (const float*)d_in[8];
  const float* bhe  = (const float*)d_in[9];
  const float* wq   = (const float*)d_in[10];
  const float* bq   = (const float*)d_in[11];
  const float* eh0  = (const float*)d_in[12];
  const float* dh0  = (const float*)d_in[13];
  float* out = (float*)d_out;

  // workspace: gi_dec then gi_enc, f16 [512][256][384] each (~96MB each)
  h16* gid = (h16*)d_ws;
  h16* gie = gid + (size_t)SEQ * BATCH * G3;

  gi_precompute_kernel<<<dim3(SEQ * BATCH / 16), dim3(128), 0, stream>>>(
      x, wihd, wihe, bihd, bihe, gid, gie);
  SingleSelfAttnGRU_74122545594412_kernel<<<dim3(BATCH / 16), dim3(256),
                                            SMEM_TOTAL, stream>>>(
      gid, gie, whd, whe, bhd, bhe, wihd, wq, bq, len, eh0, dh0, out);
}
